// OnedirectionalLSTM_18829136626049
// MI455X (gfx1250) — matrix operand order repaired and checked by reading, not compiled, not run on hardware
//
#include <hip/hip_runtime.h>
#include <hip/hip_bf16.h>
#include <stdint.h>

// Problem constants (match reference)
#define B_    64
#define T_    512
#define NIN_  256
#define H_    512
#define L_    2
#define NOUT_ 256

typedef __bf16 bf16_t;
typedef __attribute__((ext_vector_type(16))) __bf16 v16bf;
typedef __attribute__((ext_vector_type(8)))  float  v8f;
typedef unsigned int u32x4 __attribute__((ext_vector_type(4)));
typedef int          i32x8 __attribute__((ext_vector_type(8)));
typedef int          i32x4 __attribute__((ext_vector_type(4)));

// LDS row padding: +8 bf16 (= 4 dwords) per row. Keeps rows 16B-aligned for
// ds_load_b128 and makes row stride = 4 (mod 64) dwords -> conflict-free banks.
#define APAD 8

union Frag { v16bf v; uint4 q[2]; };

// A-matrix fragment (16x32 bf16, ISA layout): lane<16 holds K = kb..kb+7 and
// kb+16..kb+23; lane>=16 the kb+8 variant. Caller passes kb = k0 + 8*(lane>=16).
__device__ __forceinline__ v16bf load_fragA(const bf16_t* __restrict__ base, int ld, int row, int kb) {
  Frag f;
  const bf16_t* p = base + (size_t)row * (size_t)ld + kb;
  f.q[0] = *reinterpret_cast<const uint4*>(p);
  f.q[1] = *reinterpret_cast<const uint4*>(p + 16);
  return f.v;
}

// B-matrix fragment (32x16 bf16): lane n holds column n%16; lanes 0-15 K=k0..k0+7 and
// k0+16..k0+23, lanes 16-31 the k0+8 variant (same order as A -> two b128 loads).
__device__ __forceinline__ v16bf load_fragB(const bf16_t* __restrict__ base, int ld, int col, int kb2) {
  Frag f;
  const bf16_t* p = base + (size_t)col * (size_t)ld + (kb2 & ~31) + ((kb2 & 16) >> 1);
  f.q[0] = *reinterpret_cast<const uint4*>(p);
  f.q[1] = *reinterpret_cast<const uint4*>(p + 16);
  return f.v;
}

__device__ __forceinline__ float sigm(float x) { return 1.0f / (1.0f + __expf(-x)); }

#if __has_builtin(__builtin_amdgcn_tensor_load_to_lds)
#define HAVE_TDM 1
#else
#define HAVE_TDM 0
#endif

#if HAVE_TDM
// TDM 2-D tile load: rows x rowElems bf16 tile (global row stride rowStrideElems)
// -> LDS at lds_off, padded by 4 dwords after every row (rowElems must be a power
// of two in [4..512] so pad_interval encodes exactly one row).
__device__ __forceinline__ void tdm_load_2d(uint32_t lds_off, const void* gptr,
                                            uint32_t rows, uint32_t rowElems,
                                            uint64_t rowStrideElems) {
  const uint64_t ga = (uint64_t)(uintptr_t)gptr;
  const uint32_t ivl = (uint32_t)(__builtin_ctz(rowElems >> 1) - 1); // dwords/row = rowElems/2 = 2<<ivl
  u32x4 g0;
  g0.x = 0x1u;                                                // count=1, user mode
  g0.y = lds_off;                                             // lds_addr
  g0.z = (uint32_t)ga;                                        // global_addr[31:0]
  g0.w = (uint32_t)((ga >> 32) & 0x1FFFFFFu) | (2u << 30);    // global_addr[56:32] | type=2
  i32x8 g1;
  g1[0] = (int)((1u << 16)            // data_size = 2 bytes
              | (1u << 20)            // pad_enable
              | (ivl << 22)           // pad_interval
              | (3u << 25));          // pad_amount = 4 dwords
  g1[1] = (int)((rowElems & 0xFFFFu) << 16);                          // tensor_dim0[15:0]
  g1[2] = (int)(((rowElems >> 16) & 0xFFFFu) | ((rows & 0xFFFFu) << 16)); // dim0 hi | tensor_dim1 lo
  g1[3] = (int)(((rows >> 16) & 0xFFFFu) | ((rowElems & 0xFFFFu) << 16)); // dim1 hi | tile_dim0
  g1[4] = (int)(rows & 0xFFFFu);                                      // tile_dim1 (tile_dim2 = 0)
  g1[5] = (int)(uint32_t)(rowStrideElems & 0xFFFFFFFFu);              // tensor_dim0_stride[31:0]
  g1[6] = (int)(uint32_t)((rowStrideElems >> 32) & 0xFFFFu);          // stride[47:32] (dim1_stride = 0)
  g1[7] = 0;
  const i32x4 z4 = {0, 0, 0, 0};
#if defined(__clang_major__) && (__clang_major__ >= 23)
  const i32x8 z8 = {0, 0, 0, 0, 0, 0, 0, 0};
  __builtin_amdgcn_tensor_load_to_lds(g0, g1, z4, z4, z8, 0);
#else
  __builtin_amdgcn_tensor_load_to_lds(g0, g1, z4, z4, 0);
#endif
}
#else
// Fallback: cooperative vector copy global -> padded LDS.
__device__ __forceinline__ void coop_copy_2d(bf16_t* dst, const bf16_t* src,
                                             int rows, int rowElems, int ldsrc, int tid) {
  const int chunksPerRow = rowElems >> 3;  // uint4 = 8 bf16
  for (int c = tid; c < rows * chunksPerRow; c += 256) {
    const int r = c / chunksPerRow, cc = c - r * chunksPerRow;
    *reinterpret_cast<uint4*>(dst + (size_t)r * (rowElems + APAD) + cc * 8) =
        *reinterpret_cast<const uint4*>(src + (size_t)r * ldsrc + cc * 8);
  }
}
#endif

// Fused LSTM layer step: TDM-stage activations to LDS, gates GEMM (WMMA bf16->f32),
// bias, cell update. Grid: 32 blocks (one per 16-wide H slice), 256 threads (8 waves).
__global__ __launch_bounds__(256)
void lstm_layer_step(const bf16_t* __restrict__ Ax, int ldax,          // default input [64 x K1]
                     const bf16_t* __restrict__ Aprev, int ldaprev,    // scheduled-sampling input
                     const uint8_t* __restrict__ use_prev, int t,
                     const bf16_t* __restrict__ Hprev,                 // [64 x 512] bf16
                     const bf16_t* __restrict__ Wi, int K1,            // [2048 x K1] bf16
                     const bf16_t* __restrict__ Wh,                    // [2048 x 512] bf16
                     const float*  __restrict__ bsum,                  // [2048] (b_ih + b_hh)
                     float*        __restrict__ cst,                   // [64 x 512] f32 (in-place)
                     bf16_t*       __restrict__ Hnext)                 // [64 x 512] bf16
{
  extern __shared__ char smem[];
  bf16_t* Axl = reinterpret_cast<bf16_t*>(smem);                              // [64][K1+APAD]
  bf16_t* Ahl = Axl + (size_t)B_ * (K1 + APAD);                               // [64][H_+APAD]
  float*  gl  = reinterpret_cast<float*>(Ahl + (size_t)B_ * (H_ + APAD));     // [64][65]

  const int w    = threadIdx.x >> 5;
  const int lane = threadIdx.x & 31;
  const int mi   = w & 3;              // M tile (16 rows of batch)
  const int g0   = (w >> 2) << 1;      // this wave handles gates g0 and g0+1
  const int n0   = blockIdx.x << 4;    // H-slice start
  const int lrow = lane & 15;
  const int klo  = (lane >> 4) << 3;   // 0 or 8 (A-frag K offset AND C-frag row offset)
  const int kb2  = (lane >> 4) << 4;   // 0 or 16 (B-frag K offset)

  const bf16_t* A = Ax; int lda = ldax;
  if (use_prev != nullptr && use_prev[t]) { A = Aprev; lda = ldaprev; }  // uniform branch

  // ---- stage activation panels into LDS ----
#if HAVE_TDM
  if (w == 0) {
    const uint32_t base = (uint32_t)(uintptr_t)smem;
    tdm_load_2d(base, A, B_, (uint32_t)K1, (uint64_t)lda);
    tdm_load_2d(base + (uint32_t)(B_ * (K1 + APAD) * 2), Hprev, B_, H_, H_);
    __builtin_amdgcn_s_wait_tensorcnt(0);
  }
#else
  coop_copy_2d(Axl, A, B_, K1, lda, threadIdx.x);
  coop_copy_2d(Ahl, Hprev, B_, H_, H_, threadIdx.x);
#endif
  __syncthreads();

  const int arow  = mi * 16 + lrow;
  const int bcol0 = g0 * H_ + n0 + lrow;        // W row == gate column index
  const int bcol1 = bcol0 + H_;

  v8f acc0 = {0.f,0.f,0.f,0.f,0.f,0.f,0.f,0.f};
  v8f acc1 = {0.f,0.f,0.f,0.f,0.f,0.f,0.f,0.f};

  // Input contribution: gates += X @ W_ih^T  (A from LDS, B from global/L2)
  for (int k = 0; k < K1; k += 32) {
    v16bf a  = load_fragA(Axl, K1 + APAD, arow, k + klo);
    v16bf b0 = load_fragB(Wi, K1, bcol0, k + kb2);
    v16bf b1 = load_fragB(Wi, K1, bcol1, k + kb2);
    acc0 = __builtin_amdgcn_wmma_f32_16x16x32_bf16(false, a, false, b0, (short)0, acc0, false, false);
    acc1 = __builtin_amdgcn_wmma_f32_16x16x32_bf16(false, a, false, b1, (short)0, acc1, false, false);
  }
  // Hidden contribution: gates += H @ W_hh^T
  for (int k = 0; k < H_; k += 32) {
    v16bf a  = load_fragA(Ahl, H_ + APAD, arow, k + klo);
    v16bf b0 = load_fragB(Wh, H_, bcol0, k + kb2);
    v16bf b1 = load_fragB(Wh, H_, bcol1, k + kb2);
    acc0 = __builtin_amdgcn_wmma_f32_16x16x32_bf16(false, a, false, b0, (short)0, acc0, false, false);
    acc1 = __builtin_amdgcn_wmma_f32_16x16x32_bf16(false, a, false, b1, (short)0, acc1, false, false);
  }

  // Bias + stage gates to LDS (C/D layout: elem r -> row mi*16 + klo + r, col lane%16)
  const float bias0 = bsum[bcol0];
  const float bias1 = bsum[bcol1];
  const int colA = g0 * 16 + lrow;
  const int colB = colA + 16;
  const int rb   = mi * 16 + klo;
#pragma unroll
  for (int r = 0; r < 8; ++r) {
    gl[(rb + r) * 65 + colA] = acc0[r] + bias0;
    gl[(rb + r) * 65 + colB] = acc1[r] + bias1;
  }
  __syncthreads();

  // Elementwise LSTM cell update for this block's 16 H columns, all 64 batch rows.
#pragma unroll
  for (int e = threadIdx.x; e < B_ * 16; e += 256) {
    const int b  = e >> 4;
    const int jl = e & 15;
    const float ig = gl[b * 65 + jl];
    const float fg = gl[b * 65 + 16 + jl];
    const float gg = gl[b * 65 + 32 + jl];
    const float og = gl[b * 65 + 48 + jl];
    const size_t idx = (size_t)b * H_ + n0 + jl;
    const float cn = sigm(fg) * cst[idx] + sigm(ig) * tanhf(gg);
    cst[idx] = cn;
    Hnext[idx] = (bf16_t)(sigm(og) * tanhf(cn));
  }
}

// Head: out[:, t, :] = tanh(H1 @ W_emb^T + b_emb); also emit bf16 prev_out feedback.
// Grid: 4 blocks x 256 threads; each block covers 64 output columns.
__global__ __launch_bounds__(256)
void head_step(const bf16_t* __restrict__ H1,      // [64 x 512] bf16
               const bf16_t* __restrict__ Wemb,    // [256 x 512] bf16
               const float*  __restrict__ bemb,    // [256]
               float*        __restrict__ out,     // [B, T, NOUT] f32
               bf16_t*       __restrict__ prevout, // [64 x 256] bf16
               int t)
{
  const int w    = threadIdx.x >> 5;
  const int lane = threadIdx.x & 31;
  const int mi   = w & 3;
  const int n2   = (w >> 2) << 1;
  const int lrow = lane & 15;
  const int klo  = (lane >> 4) << 3;
  const int kb2  = (lane >> 4) << 4;
  const int nb   = blockIdx.x << 6;

  const int arow  = mi * 16 + lrow;
  const int ncol0 = nb + n2 * 16 + lrow;
  const int ncol1 = ncol0 + 16;

  v8f acc0 = {0.f,0.f,0.f,0.f,0.f,0.f,0.f,0.f};
  v8f acc1 = {0.f,0.f,0.f,0.f,0.f,0.f,0.f,0.f};

  for (int k = 0; k < H_; k += 32) {
    v16bf a  = load_fragA(H1,   H_, arow,  k + klo);
    v16bf b0 = load_fragB(Wemb, H_, ncol0, k + kb2);
    v16bf b1 = load_fragB(Wemb, H_, ncol1, k + kb2);
    acc0 = __builtin_amdgcn_wmma_f32_16x16x32_bf16(false, a, false, b0, (short)0, acc0, false, false);
    acc1 = __builtin_amdgcn_wmma_f32_16x16x32_bf16(false, a, false, b1, (short)0, acc1, false, false);
  }

  const float bb0 = bemb[ncol0];
  const float bb1 = bemb[ncol1];
  const int rb = mi * 16 + klo;
#pragma unroll
  for (int r = 0; r < 8; ++r) {
    const int row = rb + r;
    const float y0 = tanhf(acc0[r] + bb0);
    const float y1 = tanhf(acc1[r] + bb1);
    out[(size_t)row * (T_ * NOUT_) + (size_t)t * NOUT_ + ncol0] = y0;
    out[(size_t)row * (T_ * NOUT_) + (size_t)t * NOUT_ + ncol1] = y1;
    prevout[row * NOUT_ + ncol0] = (bf16_t)y0;
    prevout[row * NOUT_ + ncol1] = (bf16_t)y1;
  }
}

// ---- prep kernels ----
__global__ void cvt_f32_bf16(const float* __restrict__ s, bf16_t* __restrict__ d, size_t n) {
  size_t i = (size_t)blockIdx.x * blockDim.x + threadIdx.x;
  const size_t stride = (size_t)gridDim.x * blockDim.x;
  for (; i < n; i += stride) d[i] = (bf16_t)s[i];
}

__global__ void add_f32(const float* __restrict__ a, const float* __restrict__ b,
                        float* __restrict__ d, int n) {
  const int i = blockIdx.x * blockDim.x + threadIdx.x;
  if (i < n) d[i] = a[i] + b[i];
}

extern "C" void kernel_launch(void* const* d_in, const int* in_sizes, int n_in,
                              void* d_out, int out_size, void* d_ws, size_t ws_size,
                              hipStream_t stream) {
  (void)in_sizes; (void)n_in; (void)out_size; (void)ws_size;

  const float*   inputs  = (const float*)d_in[0];   // [B,T,NIN]
  const float*   h0_in   = (const float*)d_in[1];   // [L,B,H]
  const float*   c0_in   = (const float*)d_in[2];   // [L,B,H]
  const float*   W_ih0   = (const float*)d_in[3];   // [4H,NIN]
  const float*   W_ih1   = (const float*)d_in[4];   // [4H,H]
  const float*   W_hh    = (const float*)d_in[5];   // [L,4H,H]
  const float*   b_ih    = (const float*)d_in[6];   // [L,4H]
  const float*   b_hh    = (const float*)d_in[7];   // [L,4H]
  const float*   W_emb   = (const float*)d_in[8];   // [NOUT,H]
  const float*   b_emb   = (const float*)d_in[9];   // [NOUT]
  const uint8_t* use_prev= (const uint8_t*)d_in[10];// [T] bool
  float* out = (float*)d_out;

  // ---- workspace layout ----
  char* ws = (char*)d_ws;
  size_t off = 0;
  auto alloc = [&](size_t bytes) -> void* {
    void* p = ws + off;
    off = (off + bytes + 255) & ~(size_t)255;
    return p;
  };
  bf16_t* xbf     = (bf16_t*)alloc((size_t)B_ * T_ * NIN_ * 2);
  bf16_t* Wih0bf  = (bf16_t*)alloc((size_t)4 * H_ * NIN_ * 2);
  bf16_t* Wih1bf  = (bf16_t*)alloc((size_t)4 * H_ * H_ * 2);
  bf16_t* Whh0bf  = (bf16_t*)alloc((size_t)4 * H_ * H_ * 2);
  bf16_t* Whh1bf  = (bf16_t*)alloc((size_t)4 * H_ * H_ * 2);
  bf16_t* Wembbf  = (bf16_t*)alloc((size_t)NOUT_ * H_ * 2);
  float*  bsum0   = (float*)alloc((size_t)4 * H_ * 4);
  float*  bsum1   = (float*)alloc((size_t)4 * H_ * 4);
  bf16_t* h0buf[2], *h1buf[2];
  h0buf[0] = (bf16_t*)alloc((size_t)B_ * H_ * 2);
  h0buf[1] = (bf16_t*)alloc((size_t)B_ * H_ * 2);
  h1buf[0] = (bf16_t*)alloc((size_t)B_ * H_ * 2);
  h1buf[1] = (bf16_t*)alloc((size_t)B_ * H_ * 2);
  float*  c0ws    = (float*)alloc((size_t)B_ * H_ * 4);
  float*  c1ws    = (float*)alloc((size_t)B_ * H_ * 4);
  bf16_t* prevout = (bf16_t*)alloc((size_t)B_ * NOUT_ * 2);

  // ---- prep: f32 -> bf16 conversions, bias fusion, state init ----
  auto cvt = [&](const float* s, bf16_t* d, size_t n) {
    int blocks = (int)((n + 1023) / 1024);
    if (blocks > 2048) blocks = 2048;
    if (blocks < 1) blocks = 1;
    cvt_f32_bf16<<<blocks, 256, 0, stream>>>(s, d, n);
  };
  cvt(inputs, xbf,    (size_t)B_ * T_ * NIN_);
  cvt(W_ih0,  Wih0bf, (size_t)4 * H_ * NIN_);
  cvt(W_ih1,  Wih1bf, (size_t)4 * H_ * H_);
  cvt(W_hh,                         Whh0bf, (size_t)4 * H_ * H_);
  cvt(W_hh + (size_t)4 * H_ * H_,   Whh1bf, (size_t)4 * H_ * H_);
  cvt(W_emb,  Wembbf, (size_t)NOUT_ * H_);
  cvt(h0_in,                    h0buf[0], (size_t)B_ * H_);
  cvt(h0_in + (size_t)B_ * H_,  h1buf[0], (size_t)B_ * H_);
  add_f32<<<(4 * H_ + 255) / 256, 256, 0, stream>>>(b_ih,          b_hh,          bsum0, 4 * H_);
  add_f32<<<(4 * H_ + 255) / 256, 256, 0, stream>>>(b_ih + 4 * H_, b_hh + 4 * H_, bsum1, 4 * H_);
  hipMemcpyAsync(c0ws, c0_in,                    (size_t)B_ * H_ * 4, hipMemcpyDeviceToDevice, stream);
  hipMemcpyAsync(c1ws, c0_in + (size_t)B_ * H_,  (size_t)B_ * H_ * 4, hipMemcpyDeviceToDevice, stream);

  // Dynamic LDS: activation panels (padded) + gate staging buffer
  const int smem0 = (B_ * (NIN_ + APAD) + B_ * (H_ + APAD)) * 2 + B_ * 65 * 4;
  const int smem1 = (B_ * (H_  + APAD) + B_ * (H_ + APAD)) * 2 + B_ * 65 * 4;
  hipFuncSetAttribute((const void*)lstm_layer_step,
                      hipFuncAttributeMaxDynamicSharedMemorySize,
                      smem1 > smem0 ? smem1 : smem0);

  // ---- sequential scan over T ----
  int cur = 0;
  for (int t = 0; t < T_; ++t) {
    const int nxt = cur ^ 1;
    // Layer 0: input is x_t (row stride T*NIN) or prev_out (scheduled sampling)
    lstm_layer_step<<<H_ / 16, 256, smem0, stream>>>(
        xbf + (size_t)t * NIN_, T_ * NIN_,
        prevout, NOUT_,
        use_prev, t,
        h0buf[cur],
        Wih0bf, NIN_,
        Whh0bf,
        bsum0,
        c0ws,
        h0buf[nxt]);
    // Layer 1: input is layer0's fresh hidden state
    lstm_layer_step<<<H_ / 16, 256, smem1, stream>>>(
        h0buf[nxt], H_,
        nullptr, 0,
        nullptr, t,
        h1buf[cur],
        Wih1bf, H_,
        Whh1bf,
        bsum1,
        c1ws,
        h1buf[nxt]);
    // Head: tanh(Linear) -> out[:, t, :] and bf16 feedback buffer
    head_step<<<NOUT_ / 64, 256, 0, stream>>>(
        h1buf[nxt], Wembbf, b_emb, out, prevout, t);
    cur = nxt;
  }
}
